// MaskDiffusion_64819646431739
// MI455X (gfx1250) — compile-verified, matrix operand order checked by reading, not run on hardware
//
#include <hip/hip_runtime.h>
#include <hip/hip_bf16.h>

// BD3-LM masked-diffusion reverse step for MI455X (gfx1250).
// One workgroup per (b,s) row; exp-values live in LDS (125KB of 320KB/WGP).
// TDM loads the logits row, WMMA evaluates 16 nucleus thresholds per pass.

#define MASK_ID_C   31999
#define NEG_INF_C   (-1000000.0f)
#define P_NUC       0.9f
#define DT_C        0.05f
#define BS          512
#define NW          (BS / 32)      // 16 waves (wave32)

typedef __attribute__((ext_vector_type(16))) _Float16     v16h;
typedef __attribute__((ext_vector_type(8)))  float        v8f;
typedef __attribute__((ext_vector_type(4)))  unsigned int v4u;
typedef __attribute__((ext_vector_type(8)))  int          v8i;
typedef __attribute__((ext_vector_type(4)))  int          v4i;

__device__ __forceinline__ float blockSum(float v, float* scr) {
    const int lane = threadIdx.x & 31;
    const int w    = threadIdx.x >> 5;
#pragma unroll
    for (int o = 16; o; o >>= 1) v += __shfl_xor(v, o, 32);
    if (lane == 0) scr[w] = v;
    __syncthreads();
    float r;
    if (w == 0) {
        r = (lane < NW) ? scr[lane] : 0.0f;
#pragma unroll
        for (int o = 16; o; o >>= 1) r += __shfl_xor(r, o, 32);
        if (lane == 0) scr[0] = r;
    }
    __syncthreads();
    r = scr[0];
    __syncthreads();
    return r;
}

__device__ __forceinline__ float blockMax(float v, float* scr) {
    const int lane = threadIdx.x & 31;
    const int w    = threadIdx.x >> 5;
#pragma unroll
    for (int o = 16; o; o >>= 1) v = fmaxf(v, __shfl_xor(v, o, 32));
    if (lane == 0) scr[w] = v;
    __syncthreads();
    float r;
    if (w == 0) {
        r = (lane < NW) ? scr[lane] : -3.4e38f;
#pragma unroll
        for (int o = 16; o; o >>= 1) r = fmaxf(r, __shfl_xor(r, o, 32));
        if (lane == 0) scr[0] = r;
    }
    __syncthreads();
    r = scr[0];
    __syncthreads();
    return r;
}

__global__ __launch_bounds__(BS, 1)
void MaskDiffusion_64819646431739_kernel(const int* __restrict__ x,
                                         const float* __restrict__ logits,
                                         const float* __restrict__ tptr,
                                         const float* __restrict__ u,
                                         float* __restrict__ xOut,
                                         float* __restrict__ pOut,
                                         int V) {
    extern __shared__ float dynsmem[];
    float* eBuf = dynsmem;          // V floats: logits, then exp-values
    float* Fscr = dynsmem + V;      // 256 floats: per-wave WMMA partials
    float* scr  = Fscr + 256;       // 64 floats: reductions / argmax
    float* bc   = scr + 64;         // 8 floats: search-interval broadcast

    const int row  = blockIdx.x;
    const int tid  = threadIdx.x;
    const int lane = tid & 31;
    const int w    = tid >> 5;

    const float* lrow = logits + (size_t)row * V;
    const float* urow = u      + (size_t)row * V;
    float*       prow = pOut   + (size_t)row * V;

    // ---- Phase 0: DMA logits row into LDS via the Tensor Data Mover ----
#if __has_builtin(__builtin_amdgcn_tensor_load_to_lds) && __has_builtin(__builtin_amdgcn_s_wait_tensorcnt)
    if (w == 0) {
        unsigned long long ga = (unsigned long long)(uintptr_t)lrow;
        unsigned lds0 = (unsigned)(uintptr_t)eBuf;
        v4u g0;
        g0[0] = 1u;                                              // count=1, user desc
        g0[1] = lds0;                                            // lds_addr
        g0[2] = (unsigned)(ga & 0xFFFFFFFFull);                  // global_addr[31:0]
        g0[3] = (unsigned)((ga >> 32) & 0x01FFFFFFull) | (2u << 30); // addr[56:32] | type=2
        v8i g1;
        g1[0] = (int)(2u << 16);                                 // data_size = 4B
        g1[1] = (int)(((unsigned)V & 0xFFFFu) << 16);            // tensor_dim0[15:0]
        g1[2] = (int)((((unsigned)V >> 16) & 0xFFFFu) | (1u << 16)); // dim0 hi | tensor_dim1=1
        g1[3] = (int)(((unsigned)V & 0xFFFFu) << 16);            // tile_dim0 = V (1D tile)
        g1[4] = 0;                                               // tile_dim1/2 unused
        g1[5] = (int)(unsigned)V;                                // tensor_dim0_stride
        g1[6] = 0;
        g1[7] = 0;
        v4i gz  = {0, 0, 0, 0};
        v8i gz8 = {0, 0, 0, 0, 0, 0, 0, 0};
        __builtin_amdgcn_tensor_load_to_lds(g0, g1, gz, gz, gz8, 0);
        __builtin_amdgcn_s_wait_tensorcnt((short)0);
    }
#else
    for (int i = tid; i < V; i += BS) eBuf[i] = lrow[i];
#endif
    __syncthreads();

    // ---- Phase 1: row max (logit at MASK_ID forced to NEG_INF) ----
    float lmax = -3.4e38f;
    for (int i = tid; i < V; i += BS) {
        float l = (i == MASK_ID_C) ? NEG_INF_C : eBuf[i];
        lmax = fmaxf(lmax, l);
    }
    float m = blockMax(lmax, scr);

    // ---- Phase 2: e_i = exp(l_i - m) into LDS; Z = sum ----
    float lsum = 0.0f;
    for (int i = tid; i < V; i += BS) {
        float l = (i == MASK_ID_C) ? NEG_INF_C : eBuf[i];
        float e = __expf(l - m);
        eBuf[i] = e;
        lsum += e;
    }
    float Z  = blockSum(lsum, scr);
    float pZ = P_NUC * Z;

    // ---- Phase 3: nucleus threshold tau: keep i iff e_i >= tau ----
    // keep(i) <=> F(e_i) <= p*Z where F(t) = sum_{e_j >= t} e_j (top-1 is
    // e=1.0 and hi<=1.0 always, so top-1 stays kept automatically).
    float lo = 0.0f, hi = 1.0f;
#if __has_builtin(__builtin_amdgcn_wmma_f32_16x16x32_f16)
    // 16-ary search: one WMMA evaluates F at 16 thresholds per 32-elem chunk.
    v16h bOnes;
#pragma unroll
    for (int s2 = 0; s2 < 16; ++s2) bOnes[s2] = (_Float16)1.0f;
    const int kbase = (lane >> 4) << 3;   // K offset 0 / 8 (A-matrix half)
    const int CH    = V >> 5;             // full 32-wide chunks
    const int tail  = CH << 5;
    for (int pass = 0; pass < 8; ++pass) {
        float wdt  = (hi - lo) * 0.0625f;
        float tauL = lo + wdt * (float)((lane & 15) + 1);  // row M = lane%16
        v8f acc = {};
        for (int c = w; c < CH; c += NW) {
            const float* ep = eBuf + (c << 5);
            v16h a;
#pragma unroll
            for (int s = 0; s < 16; ++s) {
                int vv = s >> 1, hb = s & 1;
                int K  = ((vv < 4) ? (vv * 2 + hb) : (16 + (vv - 4) * 2 + hb)) + kbase;
                float ev = ep[K];
                a[s] = (ev >= tauL) ? (_Float16)ev : (_Float16)0.0f;
            }
            acc = __builtin_amdgcn_wmma_f32_16x16x32_f16(false, a, false, bOnes,
                                                         (short)0, acc, false, false);
        }
        // D layout: lane0 d[j] = F(tau_j), lane16 d[j] = F(tau_{8+j})
        if ((lane & 15) == 0) {
            int off = w * 16 + kbase;
#pragma unroll
            for (int j = 0; j < 8; ++j) Fscr[off + j] = acc[j];
        }
        __syncthreads();
        if (w == 0) {
            float Fk = 0.0f;
            if (lane < 16) {
                for (int ww = 0; ww < NW; ++ww) Fk += Fscr[ww * 16 + lane];
                float tk = lo + wdt * (float)(lane + 1);
                for (int i = tail; i < V; ++i) {       // scalar tail (V%32)
                    float ev = eBuf[i];
                    if (ev >= tk) Fk += ev;
                }
            }
            unsigned long long mb = __ballot((lane < 16) && (Fk <= pZ));
            if (lane == 0) {
                mb &= 0xFFFFull;
                if (mb) {
                    int k = (int)__ffsll(mb) - 1;      // first threshold with F<=pZ
                    bc[1] = lo + wdt * (float)(k + 1);
                    bc[0] = (k == 0) ? lo : (lo + wdt * (float)k);
                } else {                               // f16 roundoff: shrink from below
                    bc[0] = lo + wdt * 15.0f;
                    bc[1] = hi;
                }
            }
        }
        __syncthreads();
        lo = bc[0];
        hi = bc[1];
        __syncthreads();
    }
#else
    for (int it = 0; it < 30; ++it) {
        float mid = 0.5f * (lo + hi);
        float loc = 0.0f;
        for (int i = tid; i < V; i += BS) {
            float e = eBuf[i];
            if (e >= mid) loc += e;
        }
        float F = blockSum(loc, scr);
        if (F <= pZ) hi = mid; else lo = mid;
    }
#endif
    float tau = hi;

    // ---- Phase 4: exact kept mass (f32) ----
    float ls = 0.0f;
    for (int i = tid; i < V; i += BS) {
        float e = eBuf[i];
        if (e >= tau) ls += e;
    }
    float Sk   = blockSum(ls, scr);   // >= 1.0 (max element always kept)
    float invS = 1.0f / Sk;

    float tt        = tptr[0];
    float mask_prob = (tt - DT_C) / tt;   // ((1-eps)(t-dt)) / ((1-eps)t)
    float om        = 1.0f - mask_prob;

    // ---- Phase 5: write p_x0; Gumbel-max sample over q_xs ----
    float best = -3.4e38f;
    int   bidx = 2147483647;
    for (int i = tid; i < V; i += BS) {
        float e  = eBuf[i];
        float pk = (e >= tau) ? e * invS : 0.0f;
        prow[i] = pk;
        float q  = (i == MASK_ID_C) ? mask_prob : pk * om;
        float uu = urow[i];
        float g  = -__logf(-__logf(uu + 1e-10f) + 1e-10f);
        float sc = q / (g + 1e-10f);
        if (sc > best || (sc == best && i < bidx)) { best = sc; bidx = i; }
    }
#pragma unroll
    for (int o = 16; o; o >>= 1) {
        float ov = __shfl_xor(best, o, 32);
        int   oi = __shfl_xor(bidx, o, 32);
        if (ov > best || (ov == best && oi < bidx)) { best = ov; bidx = oi; }
    }
    int* iscr = (int*)(scr + 32);
    if (lane == 0) { scr[w] = best; iscr[w] = bidx; }
    __syncthreads();
    if (tid == 0) {
        float bv = scr[0];
        int   bi = iscr[0];
        for (int ww = 1; ww < NW; ++ww) {
            float ov = scr[ww];
            int   oi = iscr[ww];
            if (ov > bv || (ov == bv && oi < bi)) { bv = ov; bi = oi; }
        }
        int xv = x[row];
        xOut[row] = (xv == MASK_ID_C) ? (float)bi : (float)xv;
    }
}

extern "C" void kernel_launch(void* const* d_in, const int* in_sizes, int n_in,
                              void* d_out, int out_size, void* d_ws, size_t ws_size,
                              hipStream_t stream) {
    const int*   x      = (const int*)d_in[0];     // [B,S] int32
    const float* logits = (const float*)d_in[1];   // [B,S,V] f32
    const float* t      = (const float*)d_in[2];   // [1] f32
    const float* u      = (const float*)d_in[3];   // [B,S,V] f32
    const int rows = in_sizes[0];                  // B*S
    const int V    = in_sizes[1] / rows;           // vocab

    float* out  = (float*)d_out;                   // [rows] x_new ++ [rows*V] p_x0
    float* xOut = out;
    float* pOut = out + rows;

    const size_t shmem = (size_t)V * sizeof(float) + 2048;
    MaskDiffusion_64819646431739_kernel<<<dim3(rows), dim3(BS), shmem, stream>>>(
        x, logits, t, u, xOut, pOut, V);
}
